// down_22574348108540
// MI455X (gfx1250) — matrix operand order repaired and checked by reading, not compiled, not run on hardware
//
#include <hip/hip_runtime.h>

typedef unsigned short ushort_t;
typedef unsigned int uint_t;

typedef __attribute__((ext_vector_type(16))) __bf16 v16bf;
typedef __attribute__((ext_vector_type(8)))  float  v8f;
typedef __attribute__((ext_vector_type(4)))  float  v4f;
typedef __attribute__((ext_vector_type(4)))  uint_t v4u;
typedef __attribute__((ext_vector_type(2)))  uint_t v2u;

struct U16x8  { ushort_t d[8]; };
struct U16x4  { ushort_t d[4]; };
struct F32x8  { float d[8]; };
struct V4U2   { v4u x, y; };

__device__ __forceinline__ ushort_t f2bf(float f) {
  // round-to-nearest-even fp32 -> bf16
  uint_t u = __builtin_bit_cast(uint_t, f);
  u += 0x7FFFu + ((u >> 16) & 1u);
  return (ushort_t)(u >> 16);
}
__device__ __forceinline__ float bf2f(ushort_t h) {
  return __builtin_bit_cast(float, (uint_t)h << 16);
}

// ---------------------------------------------------------------------------
// Kernel: zero the stats scratch (deterministic per launch)
// ---------------------------------------------------------------------------
__global__ void k_zero(float* __restrict__ p, int n) {
  int i = blockIdx.x * 256 + threadIdx.x;
  if (i < n) p[i] = 0.f;
}

// ---------------------------------------------------------------------------
// Kernel: conv1 3->64, direct (memory bound, K=27), + bias, stats accumulation
// grid (4096, 64), block 256. Writes pre-BN bf16.
// ---------------------------------------------------------------------------
__global__ void k_conv1(const float* __restrict__ x, const float* __restrict__ w1,
                        const float* __restrict__ b1, ushort_t* __restrict__ y1,
                        float* __restrict__ stats) {
  const int HW = 512 * 512;
  int tid = threadIdx.x;
  int co  = blockIdx.y;
  int p   = blockIdx.x * 256 + tid;   // over N*HW = 1,048,576
  int n   = p >> 18;
  int hw  = p & (HW - 1);
  int h   = hw >> 9, w = hw & 511;

  float acc = b1[co];
#pragma unroll
  for (int ci = 0; ci < 3; ++ci) {
    const float* xs = x  + (n * 3 + ci) * HW;
    const float* ws = w1 + (co * 3 + ci) * 9;
#pragma unroll
    for (int dh = -1; dh <= 1; ++dh) {
      int hh = h + dh;
#pragma unroll
      for (int dw = -1; dw <= 1; ++dw) {
        int ww = w + dw;
        float v = 0.f;
        if (hh >= 0 && hh < 512 && ww >= 0 && ww < 512) v = xs[hh * 512 + ww];
        acc += v * ws[(dh + 1) * 3 + (dw + 1)];
      }
    }
  }
  y1[(n * 64 + co) * HW + hw] = f2bf(acc);

  // block-level reduction: all threads in block share the same co
  __shared__ float r1[256];
  __shared__ float r2[256];
  r1[tid] = acc; r2[tid] = acc * acc;
  __syncthreads();
  for (int s = 128; s > 0; s >>= 1) {
    if (tid < s) { r1[tid] += r1[tid + s]; r2[tid] += r2[tid + s]; }
    __syncthreads();
  }
  if (tid == 0) {
    atomicAdd(&stats[co],      r1[0]);
    atomicAdd(&stats[64 + co], r2[0]);
  }
}

// ---------------------------------------------------------------------------
// Kernel: finalize BN stats -> scale/shift   (stats: [sum|sumsq|scale|shift])
// ---------------------------------------------------------------------------
__global__ void k_stats(float* __restrict__ stats, const float* __restrict__ gamma,
                        const float* __restrict__ beta, float inv_count) {
  int c = threadIdx.x;
  if (c < 64) {
    float mean = stats[c] * inv_count;
    float var  = stats[64 + c] * inv_count - mean * mean;
    float sc   = gamma[c] * rsqrtf(var + 1e-5f);
    stats[128 + c] = sc;
    stats[192 + c] = beta[c] - mean * sc;
  }
}

// ---------------------------------------------------------------------------
// Implicit-GEMM conv3x3 (CI -> 64) via v_wmma_f32_16x16x32_bf16.
// Block = 256 threads = 8 waves; tile = 64 pixels x 64 out-channels.
//   wave w: cg = w&3 (16-channel group), mg = w>>2 (32-pixel group, 2 M-tiles)
// LDS tiles are K-contiguous per row, rows padded to 16B multiples so every
// fragment is two ds_load_b128 per operand:
//   A lane needs K = kbase..kbase+7 and kbase+16..kbase+23  (kbase = 8*(l>=16))
//   B lane needs K = kb0..kb0+7 and kb0+16..kb0+23          (kb0 = 8*(l>=16))
// B fragments are shared by both M-tiles (2 WMMAs per B fetch).
// Epilogue: +bias, packed b128 store of 8 consecutive pixels, ds_add_f32 stats.
// ---------------------------------------------------------------------------
template <int CI, int HDIM, bool AFF>
__global__ void k_conv_wmma(const ushort_t* __restrict__ xin, const float* __restrict__ wgt,
                            const float* __restrict__ bias, const float* __restrict__ statsIn,
                            ushort_t* __restrict__ yout, float* __restrict__ statsOut) {
  const int HW = HDIM * HDIM;
  constexpr int L2CI = (CI == 64) ? 6 : 8;
  constexpr int LDA  = CI + 8;              // row stride (ushorts), 16B multiple
  __shared__ ushort_t Atile[64 * LDA];      // [px][ci]
  __shared__ ushort_t Btile[64 * LDA];      // [co][ci]
  __shared__ float ssum[64];
  __shared__ float ssq[64];

  int tid  = threadIdx.x;
  int lane = tid & 31;
  int wave = tid >> 5;
  int cg   = wave & 3;
  int mg   = wave >> 2;
  int base = blockIdx.x * 64;               // 64 consecutive pixels (same n)

  if (tid < 64) { ssum[tid] = 0.f; ssq[tid] = 0.f; }
  __builtin_prefetch(wgt + tid, 0, 3);      // global_prefetch_b8 on weights

  const float* sIn = statsIn + 128;         // BN scale of input tensor (AFF)
  const float* hIn = statsIn + 192;         // BN shift of input tensor (AFF)

  // Per-thread fixed A-staging coordinates (256 = 4 * 64 => px fixed).
  int pxA  = tid & 63;
  int ciA0 = tid >> 6;                      // 0..3, step 4
  int pA   = base + pxA;
  int nA   = pA / HW;
  int hwA  = pA - nA * HW;
  int hA   = hwA / HDIM, wA = hwA - hA * HDIM;
  const ushort_t* xbase = xin + (size_t)nA * CI * HW;

  // Fragment row offsets (fixed per lane).
  int l15    = lane & 15;
  int hiHalf = (lane >= 16) ? 1 : 0;
  int rowA0  = (mg * 32 + l15) * LDA;       // M-tile 0
  int rowA1  = (mg * 32 + 16 + l15) * LDA;  // M-tile 1
  int rowB   = (cg * 16 + l15) * LDA;
  int kbase  = hiHalf * 8;                  // A low-run start
  int kb0    = hiHalf * 8;                  // B run start

  v8f acc0 = {}, acc1 = {};

  for (int tap = 0; tap < 9; ++tap) {
    int dh = tap / 3 - 1, dw = tap % 3 - 1;
    __syncthreads();   // previous K-step reads done (also covers ssum init)

    // --- stage A: im2col tile, input BN folded, zero-pad outside image -----
    {
      int hh = hA + dh, ww = wA + dw;
      bool ok = (hh >= 0) & (hh < HDIM) & (ww >= 0) & (ww < HDIM);
      const ushort_t* src = xbase + hh * HDIM + ww;
      ushort_t* dst = &Atile[pxA * LDA];
#pragma unroll 4
      for (int ci = ciA0; ci < CI; ci += 4) {
        float v = 0.f;
        if (ok) {
          v = bf2f(src[ci * HW]);
          if (AFF) v = v * sIn[ci] + hIn[ci];
        }
        dst[ci] = f2bf(v);
      }
    }
    // --- stage B: weights reordered to [co][ci] for this tap ---------------
    for (int idx = tid; idx < CI * 64; idx += 256) {
      int ci = idx & (CI - 1);
      int co = idx >> L2CI;
      Btile[co * LDA + ci] = f2bf(wgt[(co * CI + ci) * 9 + tap]);
    }
    __syncthreads();

    // --- CI/32 WMMA K-steps over this tap ----------------------------------
#pragma unroll
    for (int s = 0; s < CI / 32; ++s) {
      int ks = s * 32;
      // B fragment: two contiguous 8-element K-runs (two b128 loads), shared by both tiles
      v4u bl = *(const v4u*)&Btile[rowB + ks + kb0];
      v4u bh = *(const v4u*)&Btile[rowB + ks + kb0 + 16];
      v16bf b = __builtin_bit_cast(v16bf, V4U2{bl, bh});
      // A fragments: two contiguous 8-element K-runs each (two b128 loads)
      v4u a0l = *(const v4u*)&Atile[rowA0 + ks + kbase];
      v4u a0h = *(const v4u*)&Atile[rowA0 + ks + kbase + 16];
      v16bf a0 = __builtin_bit_cast(v16bf, V4U2{a0l, a0h});
      acc0 = __builtin_amdgcn_wmma_f32_16x16x32_bf16(false, a0, false, b,
                                                     (short)0, acc0, false, false);
      v4u a1l = *(const v4u*)&Atile[rowA1 + ks + kbase];
      v4u a1h = *(const v4u*)&Atile[rowA1 + ks + kbase + 16];
      v16bf a1 = __builtin_bit_cast(v16bf, V4U2{a1l, a1h});
      acc1 = __builtin_amdgcn_wmma_f32_16x16x32_bf16(false, a1, false, b,
                                                     (short)0, acc1, false, false);
    }
  }

  // --- epilogue: bias, packed pre-BN bf16 stores, per-channel stats --------
  int   co = cg * 16 + l15;                 // every D element of this lane: N=co
  float bi = bias[co];
  float lsum = 0.f, lsq = 0.f;
#pragma unroll
  for (int mt = 0; mt < 2; ++mt) {
    F32x8 cr = __builtin_bit_cast(F32x8, mt ? acc1 : acc0);
    // lane's 8 D elements = 8 consecutive pixels of channel co
    int p0 = base + mg * 32 + mt * 16 + hiHalf * 8;
    int n  = p0 / HW;
    int hw = p0 - n * HW;
    U16x8 pk;
#pragma unroll
    for (int r = 0; r < 8; ++r) {
      float v = cr.d[r] + bi;
      pk.d[r] = f2bf(v);
      lsum += v; lsq += v * v;
    }
    *(v4u*)&yout[(size_t)(n * 64 + co) * HW + hw] = __builtin_bit_cast(v4u, pk);
  }
  atomicAdd(&ssum[co], lsum);   // ds_add_f32
  atomicAdd(&ssq[co],  lsq);
  __syncthreads();
  if (tid < 64) {
    atomicAdd(&statsOut[tid],      ssum[tid]);   // global_atomic_add_f32
    atomicAdd(&statsOut[64 + tid], ssq[tid]);
  }
}

// ---------------------------------------------------------------------------
// Kernel: BN2 + ReLU + skip_con store (fp32) + Haar DWT -> 256ch bf16
// One thread per (n, c, oh, ow-pair): b64 input loads, b128 skip stores,
// b32 per-band dwt stores.
// ---------------------------------------------------------------------------
__global__ void k_bn_relu_dwt(const ushort_t* __restrict__ y2, const float* __restrict__ stats2,
                              float* __restrict__ skip, ushort_t* __restrict__ dwt) {
  int t  = blockIdx.x * 256 + threadIdx.x;  // over 4*64*256*128
  int op = t & 127;                         // ow pair: ow = 2*op, 2*op+1
  int oh = (t >> 7) & 255;
  int c  = (t >> 15) & 63;
  int n  = t >> 21;
  float sc = stats2[128 + c], sh = stats2[192 + c];

  int ib = ((n * 64 + c) * 512 + oh * 2) * 512 + op * 4;   // 4 input cols
  U16x4 r0 = __builtin_bit_cast(U16x4, *(const v2u*)&y2[ib]);
  U16x4 r1 = __builtin_bit_cast(U16x4, *(const v2u*)&y2[ib + 512]);
  float t0[4], t1[4];
#pragma unroll
  for (int j = 0; j < 4; ++j) {
    t0[j] = fmaxf(bf2f(r0.d[j]) * sc + sh, 0.f);
    t1[j] = fmaxf(bf2f(r1.d[j]) * sc + sh, 0.f);
  }
  *(v4f*)&skip[ib]       = v4f{t0[0], t0[1], t0[2], t0[3]};
  *(v4f*)&skip[ib + 512] = v4f{t1[0], t1[1], t1[2], t1[3]};

  int ob = ((n * 256 + c) * 256 + oh) * 256 + op * 2;      // 2 output cols
  uint_t pkA = 0, pkH = 0, pkV = 0, pkD = 0;
#pragma unroll
  for (int j = 0; j < 2; ++j) {
    float a00 = t0[2 * j], a01 = t0[2 * j + 1];
    float a10 = t1[2 * j], a11 = t1[2 * j + 1];
    pkA |= (uint_t)f2bf((a00 + a01 + a10 + a11) * 0.5f) << (16 * j);
    pkH |= (uint_t)f2bf((a00 + a01 - a10 - a11) * 0.5f) << (16 * j);
    pkV |= (uint_t)f2bf((a00 - a01 + a10 - a11) * 0.5f) << (16 * j);
    pkD |= (uint_t)f2bf((a00 - a01 - a10 + a11) * 0.5f) << (16 * j);
  }
  *(uint_t*)&dwt[ob]               = pkA;   // band stride = 64*65536
  *(uint_t*)&dwt[ob + 4194304]     = pkH;
  *(uint_t*)&dwt[ob + 2 * 4194304] = pkV;
  *(uint_t*)&dwt[ob + 3 * 4194304] = pkD;
}

// ---------------------------------------------------------------------------
// Kernel: BN3 + ReLU -> dwt_out (fp32), 4 elements/thread (b64 in, b128 out)
// ---------------------------------------------------------------------------
__global__ void k_bn3_relu(const ushort_t* __restrict__ y3, const float* __restrict__ stats3,
                           float* __restrict__ out) {
  int idx = (blockIdx.x * 256 + threadIdx.x) * 4;   // over 16,777,216
  int c = (idx >> 16) & 63;                          // constant across the 4
  float sc = stats3[128 + c], sh = stats3[192 + c];
  U16x4 r = __builtin_bit_cast(U16x4, *(const v2u*)&y3[idx]);
  v4f o;
#pragma unroll
  for (int j = 0; j < 4; ++j) o[j] = fmaxf(bf2f(r.d[j]) * sc + sh, 0.f);
  *(v4f*)&out[idx] = o;
}

// ---------------------------------------------------------------------------
extern "C" void kernel_launch(void* const* d_in, const int* in_sizes, int n_in,
                              void* d_out, int out_size, void* d_ws, size_t ws_size,
                              hipStream_t stream) {
  (void)in_sizes; (void)n_in; (void)out_size; (void)ws_size;
  const float* x   = (const float*)d_in[0];
  const float* w1  = (const float*)d_in[1];
  const float* b1  = (const float*)d_in[2];
  const float* g1  = (const float*)d_in[3];
  const float* be1 = (const float*)d_in[4];
  const float* w2  = (const float*)d_in[5];
  const float* b2  = (const float*)d_in[6];
  const float* g2  = (const float*)d_in[7];
  const float* be2 = (const float*)d_in[8];
  const float* w3  = (const float*)d_in[9];
  const float* b3  = (const float*)d_in[10];
  const float* g3  = (const float*)d_in[11];
  const float* be3 = (const float*)d_in[12];

  // workspace: two 134MB bf16 buffers (reused) + stats
  ushort_t* bufA = (ushort_t*)d_ws;        // y1 (conv1 out), later dwt
  ushort_t* bufB = bufA + 67108864;        // y2 (conv2 out), later y3
  float* stats  = (float*)(bufB + 67108864);
  float* stats1 = stats;
  float* stats2 = stats + 256;
  float* stats3 = stats + 512;

  float* dwt_out = (float*)d_out;          // (4,64,256,256) fp32
  float* skip    = dwt_out + 16777216;     // (4,64,512,512) fp32

  k_zero<<<3, 256, 0, stream>>>(stats, 768);

  dim3 gc1(4096, 64);
  k_conv1<<<gc1, 256, 0, stream>>>(x, w1, b1, bufA, stats1);
  k_stats<<<1, 64, 0, stream>>>(stats1, g1, be1, 1.0f / 1048576.0f);

  k_conv_wmma<64, 512, true><<<16384, 256, 0, stream>>>(bufA, w2, b2, stats1, bufB, stats2);
  k_stats<<<1, 64, 0, stream>>>(stats2, g2, be2, 1.0f / 1048576.0f);

  k_bn_relu_dwt<<<32768, 256, 0, stream>>>(bufB, stats2, skip, bufA);

  k_conv_wmma<256, 256, false><<<4096, 256, 0, stream>>>(bufA, w3, b3, stats3, bufB, stats3);
  k_stats<<<1, 64, 0, stream>>>(stats3, g3, be3, 1.0f / 262144.0f);

  k_bn3_relu<<<16384, 256, 0, stream>>>(bufB, stats3, dwt_out);
}
